// Attention_76484777607813
// MI455X (gfx1250) — matrix operand order repaired and checked by reading, not compiled, not run on hardware
//
#include <hip/hip_runtime.h>
#include <hip/hip_bf16.h>

// ---------------------------------------------------------------------------
// Multi-head attention forward on gfx1250 (MI455X), bf16 WMMA + flash softmax.
//   B=2, N=2048, C=1024, H=16, D=64, scale = C^-0.5 = 1/32 (folded into Q).
// All WMMA operand fragments are contiguous 2x16B loads (K-major staging);
// K tiles in attention are moved by the Tensor Data Mover when available.
// ---------------------------------------------------------------------------

typedef __bf16 bf16;
typedef __attribute__((ext_vector_type(16))) __bf16          v16bf;
typedef __attribute__((ext_vector_type(8)))  float           v8f;
typedef __attribute__((ext_vector_type(8)))  unsigned short  u16x8;
typedef __attribute__((ext_vector_type(4)))  unsigned int    u32x4;
typedef __attribute__((ext_vector_type(8)))  int             i32x8;
typedef __attribute__((ext_vector_type(4)))  int             i32x4;

#define BATCH   2
#define SEQ     2048
#define EMBED   1024
#define HEADS   16
#define DHEAD   64
#define TOKENS  (BATCH * SEQ)          // 4096
#define QKVCOLS (3 * EMBED)            // 3072

#if defined(__has_builtin)
#  if __has_builtin(__builtin_amdgcn_tensor_load_to_lds)
#    define HAVE_TDM 1
#  else
#    define HAVE_TDM 0
#  endif
#else
#  define HAVE_TDM 0
#endif

// ---- fragment helpers (contiguous k order, shared by A and B) ----------------------

// Two 16-byte chunks -> one 16-element bf16 fragment (forces ds_load_b128).
__device__ __forceinline__ v16bf ld2x8(const bf16* p0, const bf16* p1) {
    union { uint4 u[2]; v16bf v; } t;
    t.u[0] = *(const uint4*)p0;
    t.u[1] = *(const uint4*)p1;
    return t.v;
}

// A-matrix 16x32 bf16 (row-major, row stride ld): lane holds row r = lane%16;
// slot e -> K = e + (lane/16)*16 (contiguous, matches frag_b_kmaj).
__device__ __forceinline__ v16bf frag_a(const bf16* p, int ld, int lane) {
    const bf16* q = p + (lane & 15) * ld + ((lane >> 4) << 4);
    return ld2x8(q, q + 8);
}

// B-matrix 32x16 (K x N) stored K-major per column: storage[n*ld + K + off].
// lane holds column n = lane%16; slot e -> K = e + (lane/16)*16 (contiguous).
__device__ __forceinline__ v16bf frag_b_kmaj(const bf16* p, int ld, int off, int lane) {
    const bf16* q = p + (lane & 15) * ld + off + ((lane >> 4) << 4);
    return ld2x8(q, q + 8);
}

__device__ __forceinline__ v8f wmma_bf16(v16bf a, v16bf b, v8f c) {
    return __builtin_amdgcn_wmma_f32_16x16x32_bf16(false, a, false, b,
                                                   (short)0, c, false, false);
}

// ---- fp32 -> bf16 conversion ----------------------------------------------

__global__ __launch_bounds__(256) void cvt_f32_bf16(const float* __restrict__ src,
                                                    bf16* __restrict__ dst, int n) {
    for (int i = blockIdx.x * blockDim.x + threadIdx.x; i < n;
         i += gridDim.x * blockDim.x)
        dst[i] = (bf16)src[i];
}

// ---- GEMM 1: qkv = x @ W_qkv, scatter into q/k/v [B,H,N,D] bf16 -----------
// Block: 256 threads (8 waves). Workgroup tile 64(M) x 128(N); wave tile 32x32.
// W tile staged transposed (K-major) so B fragments are contiguous.

__global__ __launch_bounds__(256) void gemm_qkv(const bf16* __restrict__ A,   // [4096,1024]
                                                const bf16* __restrict__ W,   // [1024,3072]
                                                bf16* __restrict__ qo,
                                                bf16* __restrict__ ko,
                                                bf16* __restrict__ vo) {
    __shared__ __align__(16) bf16 sA[64 * 32];        // [row][k]
    __shared__ __align__(16) bf16 sBt[128 * 32];      // [col][k]  (transposed)
    const int tid = threadIdx.x, lane = tid & 31, w = tid >> 5;
    const int m0 = blockIdx.y * 64, n0 = blockIdx.x * 128;
    const int wrow = (w >> 2) * 32, wcol = (w & 3) * 32;

    v8f c00 = {}, c01 = {}, c10 = {}, c11 = {};

    for (int k0 = 0; k0 < EMBED; k0 += 32) {
        // A tile: 2048 elems = 256 chunks of 8 (16B), fully coalesced.
        {
            int r = tid >> 2, c0 = (tid & 3) << 3;
            *(uint4*)&sA[tid * 8] =
                *(const uint4*)&A[(size_t)(m0 + r) * EMBED + k0 + c0];
        }
        // W tile: 4096 elems = 512 chunks; coalesced 16B global reads,
        // transposed scatter into sBt[col][k].
        unsigned short* sb = (unsigned short*)sBt;
#pragma unroll
        for (int i = 0; i < 2; ++i) {
            int cidx = tid + i * 256;
            int r = cidx >> 4, c0 = (cidx & 15) << 3;
            u16x8 wv = *(const u16x8*)&W[(size_t)(k0 + r) * QKVCOLS + n0 + c0];
#pragma unroll
            for (int j = 0; j < 8; ++j) sb[(c0 + j) * 32 + r] = wv[j];
        }
        if (k0 + 32 < EMBED)
            __builtin_prefetch(&W[(size_t)(k0 + 32) * QKVCOLS + n0 + (tid & 127)], 0, 1);
        __syncthreads();

        v16bf a0 = frag_a(&sA[(wrow + 0) * 32], 32, lane);
        v16bf a1 = frag_a(&sA[(wrow + 16) * 32], 32, lane);
        v16bf b0 = frag_b_kmaj(&sBt[(wcol + 0) * 32], 32, 0, lane);
        v16bf b1 = frag_b_kmaj(&sBt[(wcol + 16) * 32], 32, 0, lane);
        c00 = wmma_bf16(a0, b0, c00);
        c01 = wmma_bf16(a0, b1, c01);
        c10 = wmma_bf16(a1, b0, c10);
        c11 = wmma_bf16(a1, b1, c11);
        __syncthreads();
    }

    // scatter C tiles into q/k/v [B,H,N,D]; fold 1/32 softmax scale into q
#pragma unroll
    for (int i = 0; i < 2; ++i) {
#pragma unroll
        for (int j = 0; j < 2; ++j) {
            const v8f* cp = (i == 0) ? (j == 0 ? &c00 : &c01)
                                     : (j == 0 ? &c10 : &c11);
            int col = n0 + wcol + j * 16 + (lane & 15);
            int s   = col >> 10;                // 0=q 1=k 2=v
            int rem = col & 1023;
            int h   = rem >> 6, d = rem & 63;
#pragma unroll
            for (int g = 0; g < 8; ++g) {
                int row = m0 + wrow + i * 16 + g + ((lane >> 4) << 3);
                int b = row >> 11, t = row & 2047;
                size_t dst = ((size_t)(b * HEADS + h) * SEQ + t) * DHEAD + d;
                float val = (*cp)[g];
                if (s == 0)      qo[dst] = (bf16)(val * 0.03125f);
                else if (s == 1) ko[dst] = (bf16)val;
                else             vo[dst] = (bf16)val;
            }
        }
    }
}

// ---- Flash attention: per (b,h), 64 query rows per block, 4 waves ---------
// K block loaded row-major (TDM when available); V block staged transposed.

__global__ __launch_bounds__(128) void attn_flash(const bf16* __restrict__ qb,
                                                  const bf16* __restrict__ kb,
                                                  const bf16* __restrict__ vb,
                                                  bf16* __restrict__ ob) {  // [B,N,C] bf16
    __shared__ __align__(16) bf16 sK[32 * DHEAD];     // [key][d]
    __shared__ __align__(16) bf16 sVt[DHEAD * 32];    // [d][key] (transposed)
    __shared__ __align__(16) bf16 sP[4][16 * 32];
    const int tid = threadIdx.x, lane = tid & 31, w = tid >> 5;
    const int bh = blockIdx.y;                  // 0..31
    const int qrow = blockIdx.x * 64 + w * 16;  // wave's query-row base

    const bf16* qp = qb + (size_t)bh * SEQ * DHEAD;
    const bf16* kp = kb + (size_t)bh * SEQ * DHEAD;
    const bf16* vp = vb + (size_t)bh * SEQ * DHEAD;

    v16bf qa0 = frag_a(qp + (size_t)qrow * DHEAD + 0, DHEAD, lane);
    v16bf qa1 = frag_a(qp + (size_t)qrow * DHEAD + 32, DHEAD, lane);

#if HAVE_TDM
    const unsigned lds_k = (unsigned)(uintptr_t)&sK[0];
#endif

    v8f o[4] = {{}, {}, {}, {}};
    float mrow[8], lrow[8];
#pragma unroll
    for (int g = 0; g < 8; ++g) { mrow[g] = -3.0e38f; lrow[g] = 0.0f; }

    for (int k0 = 0; k0 < SEQ; k0 += 32) {
        // V tile transposed into sVt[d][key]; coalesced 16B global reads.
        unsigned short* sv = (unsigned short*)sVt;
#pragma unroll
        for (int i = 0; i < 2; ++i) {
            int cidx = tid + i * 128;          // 256 chunks of 8 over 32x64
            int r = cidx >> 3, c0 = (cidx & 7) << 3;
            u16x8 vv = *(const u16x8*)&vp[(size_t)(k0 + r) * DHEAD + c0];
#pragma unroll
            for (int j = 0; j < 8; ++j) sv[(c0 + j) * 32 + r] = vv[j];
        }

#if HAVE_TDM
        // K tile via Tensor Data Mover: 2D tile 32 rows x 64 elems, 2B each.
        if (w == 0) {
            unsigned long long ga =
                (unsigned long long)(uintptr_t)(kp + (size_t)k0 * DHEAD);
            u32x4 g0;
            g0[0] = 1u;                                   // count=1, user mode
            g0[1] = lds_k;                                // lds_addr (bytes)
            g0[2] = (unsigned)(ga & 0xffffffffu);         // global_addr[31:0]
            g0[3] = (unsigned)((ga >> 32) & 0x01ffffffu)  // global_addr[56:32]
                    | (2u << 30);                         // type = 2 (image)
            i32x8 g1;
            g1[0] = 0x00010000;          // workgroup_mask=0, data_size=1 (2B)
            g1[1] = (int)(64u << 16);    // tensor_dim0 = 64  (bits 79:48 lo)
            g1[2] = (int)(32u << 16);    // tensor_dim0 hi=0 | tensor_dim1=32 lo
            g1[3] = (int)(64u << 16);    // tensor_dim1 hi=0 | tile_dim0=64
            g1[4] = 32;                  // tile_dim1=32, tile_dim2=0
            g1[5] = 64;                  // tensor_dim0_stride = 64 (lo32)
            g1[6] = 0;                   // stride0 hi | stride1 lo
            g1[7] = 0;
            i32x4 gz = {0, 0, 0, 0};
#if defined(__clang_major__) && __clang_major__ >= 23
            i32x8 gz8 = {0, 0, 0, 0, 0, 0, 0, 0};
            __builtin_amdgcn_tensor_load_to_lds(g0, g1, gz, gz, gz8, 0);
#else
            __builtin_amdgcn_tensor_load_to_lds(g0, g1, gz, gz, 0);
#endif
            __builtin_amdgcn_s_wait_tensorcnt(0);
        }
#else
#pragma unroll
        for (int i = 0; i < 2; ++i) {
            int cidx = tid + i * 128;
            int r = cidx >> 3, c0 = (cidx & 7) << 3;
            *(uint4*)&sK[cidx * 8] =
                *(const uint4*)&kp[(size_t)(k0 + r) * DHEAD + c0];
        }
#endif
        __syncthreads();

        // S = Q * K^T  (two 16x16 key tiles; B K-dim = d, row-major sK is
        // K-major per key column -> contiguous fragments)
        v8f s0 = {}, s1 = {};
        s0 = wmma_bf16(qa0, frag_b_kmaj(sK, DHEAD, 0, lane), s0);
        s0 = wmma_bf16(qa1, frag_b_kmaj(sK, DHEAD, 32, lane), s0);
        s1 = wmma_bf16(qa0, frag_b_kmaj(sK + 16 * DHEAD, DHEAD, 0, lane), s1);
        s1 = wmma_bf16(qa1, frag_b_kmaj(sK + 16 * DHEAD, DHEAD, 32, lane), s1);

        // online softmax (rows split across half-waves: row = g + 8*(lane/16))
#pragma unroll
        for (int g = 0; g < 8; ++g) {
            float vmax = fmaxf(s0[g], s1[g]);
            vmax = fmaxf(vmax, __shfl_xor(vmax, 1));
            vmax = fmaxf(vmax, __shfl_xor(vmax, 2));
            vmax = fmaxf(vmax, __shfl_xor(vmax, 4));
            vmax = fmaxf(vmax, __shfl_xor(vmax, 8));
            float mnew = fmaxf(mrow[g], vmax);
            float corr = __expf(mrow[g] - mnew);
            mrow[g] = mnew;
            float p0 = __expf(s0[g] - mnew);
            float p1 = __expf(s1[g] - mnew);
            s0[g] = p0; s1[g] = p1;
            float rs = p0 + p1;
            rs += __shfl_xor(rs, 1);
            rs += __shfl_xor(rs, 2);
            rs += __shfl_xor(rs, 4);
            rs += __shfl_xor(rs, 8);
            lrow[g] = lrow[g] * corr + rs;
            o[0][g] *= corr; o[1][g] *= corr; o[2][g] *= corr; o[3][g] *= corr;
        }

        // transpose P through per-wave LDS scratch into A-fragment layout
#pragma unroll
        for (int g = 0; g < 8; ++g) {
            int r = g + ((lane >> 4) << 3);
            sP[w][r * 32 + (lane & 15)]      = (bf16)s0[g];
            sP[w][r * 32 + 16 + (lane & 15)] = (bf16)s1[g];
        }
        v16bf pa = frag_a(&sP[w][0], 32, lane);

        // O += P * V  (four 16-wide d tiles; sVt is K-major per d column)
#pragma unroll
        for (int dt = 0; dt < 4; ++dt)
            o[dt] = wmma_bf16(pa, frag_b_kmaj(&sVt[dt * 16 * 32], 32, 0, lane), o[dt]);

        __syncthreads();
    }

    // normalize and write [B,N,C] bf16 for the output projection
    const int b = bh >> 4, h = bh & 15;
    float inv[8];
#pragma unroll
    for (int g = 0; g < 8; ++g) inv[g] = 1.0f / lrow[g];
#pragma unroll
    for (int dt = 0; dt < 4; ++dt) {
#pragma unroll
        for (int g = 0; g < 8; ++g) {
            int r = g + ((lane >> 4) << 3);
            int t = qrow + r;
            size_t dst = ((size_t)(b * SEQ + t)) * EMBED + h * DHEAD + dt * 16 + (lane & 15);
            ob[dst] = (bf16)(o[dt][g] * inv[g]);
        }
    }
}

// ---- GEMM 2: out = attn @ W_proj + b_proj (fp32 out) ----------------------

__global__ __launch_bounds__(256) void gemm_proj(const bf16* __restrict__ A,  // [4096,1024]
                                                 const bf16* __restrict__ W,  // [1024,1024]
                                                 const float* __restrict__ bias,
                                                 float* __restrict__ out) {   // [4096,1024] f32
    __shared__ __align__(16) bf16 sA[64 * 32];
    __shared__ __align__(16) bf16 sBt[128 * 32];
    const int tid = threadIdx.x, lane = tid & 31, w = tid >> 5;
    const int m0 = blockIdx.y * 64, n0 = blockIdx.x * 128;
    const int wrow = (w >> 2) * 32, wcol = (w & 3) * 32;

    v8f c00 = {}, c01 = {}, c10 = {}, c11 = {};

    for (int k0 = 0; k0 < EMBED; k0 += 32) {
        {
            int r = tid >> 2, c0 = (tid & 3) << 3;
            *(uint4*)&sA[tid * 8] =
                *(const uint4*)&A[(size_t)(m0 + r) * EMBED + k0 + c0];
        }
        unsigned short* sb = (unsigned short*)sBt;
#pragma unroll
        for (int i = 0; i < 2; ++i) {
            int cidx = tid + i * 256;
            int r = cidx >> 4, c0 = (cidx & 15) << 3;
            u16x8 wv = *(const u16x8*)&W[(size_t)(k0 + r) * EMBED + n0 + c0];
#pragma unroll
            for (int j = 0; j < 8; ++j) sb[(c0 + j) * 32 + r] = wv[j];
        }
        if (k0 + 32 < EMBED)
            __builtin_prefetch(&W[(size_t)(k0 + 32) * EMBED + n0 + (tid & 127)], 0, 1);
        __syncthreads();

        v16bf a0 = frag_a(&sA[(wrow + 0) * 32], 32, lane);
        v16bf a1 = frag_a(&sA[(wrow + 16) * 32], 32, lane);
        v16bf b0 = frag_b_kmaj(&sBt[(wcol + 0) * 32], 32, 0, lane);
        v16bf b1 = frag_b_kmaj(&sBt[(wcol + 16) * 32], 32, 0, lane);
        c00 = wmma_bf16(a0, b0, c00);
        c01 = wmma_bf16(a0, b1, c01);
        c10 = wmma_bf16(a1, b0, c10);
        c11 = wmma_bf16(a1, b1, c11);
        __syncthreads();
    }

#pragma unroll
    for (int i = 0; i < 2; ++i) {
#pragma unroll
        for (int j = 0; j < 2; ++j) {
            const v8f* cp = (i == 0) ? (j == 0 ? &c00 : &c01)
                                     : (j == 0 ? &c10 : &c11);
            int col = n0 + wcol + j * 16 + (lane & 15);
            float bv = bias[col];
#pragma unroll
            for (int g = 0; g < 8; ++g) {
                int row = m0 + wrow + i * 16 + g + ((lane >> 4) << 3);
                out[(size_t)row * EMBED + col] = (*cp)[g] + bv;
            }
        }
    }
}

// ---- workspace layout (bytes) ---------------------------------------------
static const size_t OFF_XB    = 0;                         // 4096*1024 bf16
static const size_t OFF_WQKV  = OFF_XB    + (size_t)TOKENS * EMBED * 2;
static const size_t OFF_WPROJ = OFF_WQKV  + (size_t)EMBED * QKVCOLS * 2;
static const size_t OFF_Q     = OFF_WPROJ + (size_t)EMBED * EMBED * 2;
static const size_t OFF_K     = OFF_Q     + (size_t)TOKENS * EMBED * 2;
static const size_t OFF_V     = OFF_K     + (size_t)TOKENS * EMBED * 2;
static const size_t OFF_ATTN  = OFF_V     + (size_t)TOKENS * EMBED * 2;

extern "C" void kernel_launch(void* const* d_in, const int* in_sizes, int n_in,
                              void* d_out, int out_size, void* d_ws, size_t ws_size,
                              hipStream_t stream) {
    (void)in_sizes; (void)n_in; (void)out_size; (void)ws_size;
    const float* x     = (const float*)d_in[0];
    const float* Wqkv  = (const float*)d_in[1];
    const float* Wproj = (const float*)d_in[2];
    const float* bproj = (const float*)d_in[3];
    float* out = (float*)d_out;

    char* ws = (char*)d_ws;
    bf16* xb     = (bf16*)(ws + OFF_XB);
    bf16* wqkvb  = (bf16*)(ws + OFF_WQKV);
    bf16* wprojb = (bf16*)(ws + OFF_WPROJ);
    bf16* qb     = (bf16*)(ws + OFF_Q);
    bf16* kb     = (bf16*)(ws + OFF_K);
    bf16* vb     = (bf16*)(ws + OFF_V);
    bf16* attnb  = (bf16*)(ws + OFF_ATTN);

    cvt_f32_bf16<<<4096, 256, 0, stream>>>(x, xb, TOKENS * EMBED);
    cvt_f32_bf16<<<4096, 256, 0, stream>>>(Wqkv, wqkvb, EMBED * QKVCOLS);
    cvt_f32_bf16<<<1024, 256, 0, stream>>>(Wproj, wprojb, EMBED * EMBED);

    gemm_qkv<<<dim3(QKVCOLS / 128, TOKENS / 64), 256, 0, stream>>>(xb, wqkvb, qb, kb, vb);
    attn_flash<<<dim3(SEQ / 64, BATCH * HEADS), 128, 0, stream>>>(qb, kb, vb, attnb);
    gemm_proj<<<dim3(EMBED / 128, TOKENS / 64), 256, 0, stream>>>(attnb, wprojb, bproj, out);
}
